// mutcos_62998580298321
// MI455X (gfx1250) — compile-verified, matrix operand order checked by reading, not run on hardware
//
#include <hip/hip_runtime.h>

typedef __attribute__((ext_vector_type(16))) _Float16 v16h;
typedef __attribute__((ext_vector_type(8)))  float    v8f;

#define WMMA_F16(a, b, c) \
  __builtin_amdgcn_wmma_f32_16x16x32_f16(false, (a), false, (b), (short)0, (c), false, false)

constexpr int Bb = 8, Mm = 2048, Nn = 2048, Dd = 256, Kk = 256;

// ---------------------------------------------------------------------------
// Tensor Data Mover helpers (CDNA5 ISA/§8).
// D# group0: count=1 | lds_addr | global_addr[56:0] | type=2
// D# group1: data_size=2B, tensor_dim0=D, tensor_dim1=16 rows,
//            tile_dim0=D, tile_dim1=16, tensor_dim0_stride=D.
// ---------------------------------------------------------------------------
typedef __attribute__((ext_vector_type(4))) unsigned int tdm_v4u;
typedef __attribute__((ext_vector_type(8))) int          tdm_v8i;
typedef __attribute__((ext_vector_type(4))) int          tdm_v4i;

__device__ __forceinline__ unsigned lds_off(const void* p) {
  // generic LDS address: [63:32]=aperture, [31:0]=byte offset within LDS
  return (unsigned)(unsigned long long)p;
}

__device__ __forceinline__ void tdm_load_tile16(const _Float16* gsrc, unsigned dst_lds) {
  const unsigned long long ga = (unsigned long long)gsrc;
  tdm_v4u g0 = { 1u,                                    // count=1, user mode
                 dst_lds,                               // lds_addr (bytes)
                 (unsigned)(ga & 0xFFFFFFFFull),        // global_addr[31:0]
                 (unsigned)((ga >> 32) & 0x1FFFFFFull)  // global_addr[56:32]
                   | (2u << 30) };                      // type = 2 ("image")
  tdm_v8i g1 = { (int)(1u << 16),                       // data_size = 1 -> 2 bytes
                 (int)((unsigned)(Dd & 0xFFFF) << 16),  // tensor_dim0[15:0]=D
                 (int)((unsigned)(16 & 0xFFFF) << 16),  // tensor_dim1[15:0]=16 rows
                 (int)((unsigned)(Dd & 0xFFFF) << 16),  // tile_dim0 = D
                 (int)16,                               // tile_dim1=16, tile_dim2=0
                 (int)Dd,                               // tensor_dim0_stride[31:0]=D
                 0, 0 };                                // upper stride bits
  tdm_v4i z4 = { 0, 0, 0, 0 };                          // groups 2/3 unused (2-D tile)
#if defined(__clang_major__) && (__clang_major__ >= 23)
  tdm_v8i z8 = { 0, 0, 0, 0, 0, 0, 0, 0 };
  __builtin_amdgcn_tensor_load_to_lds(g0, g1, z4, z4, z8, 0);
#else
  __builtin_amdgcn_tensor_load_to_lds(g0, g1, z4, z4, 0);
#endif
}

// ---------------------------------------------------------------------------
// Phase 1: f32 -> f16 conversion + fused w = vu*vd
// ---------------------------------------------------------------------------
__global__ __launch_bounds__(256) void prep_kernel(
    const float* __restrict__ yu, const float* __restrict__ xu,
    const float* __restrict__ xd, const float* __restrict__ yd,
    const float* __restrict__ vu, const float* __restrict__ vd,
    _Float16* __restrict__ yu_h, _Float16* __restrict__ xu_h,
    _Float16* __restrict__ xd_h, _Float16* __restrict__ yd_h,
    _Float16* __restrict__ w_h)
{
  size_t i = (size_t)blockIdx.x * blockDim.x + threadIdx.x;
  if (i >= (size_t)Bb * Mm * Dd) return;
  yu_h[i] = (_Float16)yu[i];
  xu_h[i] = (_Float16)xu[i];
  w_h[i]  = (_Float16)(vu[i] * vd[i]);
  if (i < (size_t)Nn * Dd) {
    xd_h[i] = (_Float16)xd[i];
    yd_h[i] = (_Float16)yd[i];
  }
}

// ---------------------------------------------------------------------------
// Phase 2: column-softmax statistics over the M axis.
// 8 waves / WG; wave owns a 16-column stripe of N. A-tiles (yu/xu m-tiles,
// identical across all 8 waves) are DMA'd once per WGP by the TDM into
// double-buffered LDS and overlap with WMMA; B operands stay hoisted in VGPRs.
// ---------------------------------------------------------------------------
__global__ __launch_bounds__(256) void colstats_kernel(
    const _Float16* __restrict__ yu_h, const _Float16* __restrict__ xu_h,
    const _Float16* __restrict__ xd_h, const _Float16* __restrict__ yd_h,
    float* __restrict__ cmax_u, float* __restrict__ csum_u,
    float* __restrict__ cmax_d, float* __restrict__ csum_d)
{
  __shared__ __align__(32) _Float16 aU[2][16 * Dd];   // yu tile double-buffer
  __shared__ __align__(32) _Float16 aD[2][16 * Dd];   // xu tile double-buffer

  const int b    = blockIdx.y;
  const int wave = threadIdx.x >> 5;
  const int lane = threadIdx.x & 31;
  const int hi   = lane >> 4;
  const int ln   = lane & 15;
  const int nbase  = blockIdx.x * 128 + wave * 16;
  const int ncol   = nbase + ln;        // B-operand column for this lane
  const int dchunk = hi * 16;           // K sub-chunk per half-wave

  const _Float16* yuB = yu_h + (size_t)b * Mm * Dd;
  const _Float16* xuB = xu_h + (size_t)b * Mm * Dd;

  float rmax_u = -1e30f, rsum_u = 0.f;
  float rmax_d = -1e30f, rsum_d = 0.f;

  // prologue: DMA tile 0 into buffer 0
  if (wave == 0) {
    tdm_load_tile16(yuB, lds_off(&aU[0][0]));
    tdm_load_tile16(xuB, lds_off(&aD[0][0]));
  }

  for (int mt = 0; mt < Mm / 16; ++mt) {
    const int cur = mt & 1;
    if (wave == 0) {
      if (mt + 1 < Mm / 16) {
        // buffer cur^1 was last read in iteration mt-1, whose trailing barrier
        // already completed -> safe for the TDM to overwrite it now.
        tdm_load_tile16(yuB + (size_t)(mt + 1) * 16 * Dd, lds_off(&aU[cur ^ 1][0]));
        tdm_load_tile16(xuB + (size_t)(mt + 1) * 16 * Dd, lds_off(&aD[cur ^ 1][0]));
        __builtin_amdgcn_s_wait_tensorcnt(2);   // tile mt's two DMAs retired
      } else {
        __builtin_amdgcn_s_wait_tensorcnt(0);
      }
    }
    __syncthreads();   // release buffer `cur` to all waves

    v8f cu = {}; v8f cd = {};
#pragma unroll
    for (int ks = 0; ks < Dd / 32; ++ks) {
      const int d0 = ks * 32 + dchunk;
      v16h au = *(const v16h*)(&aU[cur][ln * Dd + d0]);
      v16h bu = *(const v16h*)(xd_h + (size_t)ncol * Dd + d0);   // hoisted to VGPRs
      cu = WMMA_F16(au, bu, cu);
      v16h ad = *(const v16h*)(&aD[cur][ln * Dd + d0]);
      v16h bd = *(const v16h*)(yd_h + (size_t)ncol * Dd + d0);   // hoisted to VGPRs
      cd = WMMA_F16(ad, bd, cd);
    }
    // ---- online column max / sum-exp for mutu ----
    {
      float tmax = cu[0];
#pragma unroll
      for (int r = 1; r < 8; ++r) tmax = fmaxf(tmax, cu[r]);
      tmax = fmaxf(tmax, __shfl_xor(tmax, 16, 32));   // combine lane-halves (M split)
      float nm = fmaxf(rmax_u, tmax);
      float ts = 0.f;
#pragma unroll
      for (int r = 0; r < 8; ++r) ts += __expf(cu[r] - nm);
      ts += __shfl_xor(ts, 16, 32);
      rsum_u = rsum_u * __expf(rmax_u - nm) + ts;
      rmax_u = nm;
    }
    // ---- online column max / sum-exp for mutd ----
    {
      float tmax = cd[0];
#pragma unroll
      for (int r = 1; r < 8; ++r) tmax = fmaxf(tmax, cd[r]);
      tmax = fmaxf(tmax, __shfl_xor(tmax, 16, 32));
      float nm = fmaxf(rmax_d, tmax);
      float ts = 0.f;
#pragma unroll
      for (int r = 0; r < 8; ++r) ts += __expf(cd[r] - nm);
      ts += __shfl_xor(ts, 16, 32);
      rsum_d = rsum_d * __expf(rmax_d - nm) + ts;
      rmax_d = nm;
    }
    __syncthreads();   // all waves done reading buffer `cur` before it is re-filled
  }

  if (lane < 16) {
    const size_t o = (size_t)b * Nn + nbase + lane;
    cmax_u[o] = rmax_u; csum_u[o] = rsum_u;
    cmax_d[o] = rmax_d; csum_d[o] = rsum_d;
  }
}

// ---------------------------------------------------------------------------
// Phase 3: fused attention + output GEMM.
// Workgroup = 64 m-rows of one batch; stream N in chunks of 32.
// Per chunk: recompute scores (WMMA), p = exp(attwu+attwd) via colstats,
// accumulate rowsum (energy in (0,2] -> no max pass needed for the N-softmax),
// stage p in LDS as f16, second WMMA chain accumulates acc[k,m].
// ---------------------------------------------------------------------------
__global__ __launch_bounds__(256) void fused_kernel(
    const _Float16* __restrict__ yu_h, const _Float16* __restrict__ xu_h,
    const _Float16* __restrict__ xd_h, const _Float16* __restrict__ yd_h,
    const _Float16* __restrict__ w_h,
    const float* __restrict__ cmax_u, const float* __restrict__ csum_u,
    const float* __restrict__ cmax_d, const float* __restrict__ csum_d,
    float* __restrict__ out)
{
  __shared__ __align__(32) _Float16 pLDS[64 * 32];  // [m_local][n_local]
  __shared__ float rowsum[64];

  const int b     = blockIdx.y;
  const int mbase = blockIdx.x * 64;
  const int wave  = threadIdx.x >> 5;
  const int lane  = threadIdx.x & 31;
  const int hi    = lane >> 4;
  const int ln    = lane & 15;
  const int mt    = wave & 3;     // score m-tile owned by this wave (0..3)
  const int nt    = wave >> 2;    // score n-tile owned by this wave (0..1)
  const int dchunk = hi * 16;

  if (threadIdx.x < 64) rowsum[threadIdx.x] = 0.f;
  __syncthreads();

  const _Float16* yuB = yu_h + (size_t)b * Mm * Dd;
  const _Float16* xuB = xu_h + (size_t)b * Mm * Dd;
  const _Float16* wB  = w_h  + (size_t)b * Kk * Nn;

  v8f acc[2][4];
#pragma unroll
  for (int i = 0; i < 2; ++i)
#pragma unroll
    for (int j = 0; j < 4; ++j) acc[i][j] = (v8f){};

  const int mrow = mbase + mt * 16 + ln;   // A-operand row (scores)

  for (int ns = 0; ns < Nn / 32; ++ns) {
    const int n0   = ns * 32;
    const int ncol = n0 + nt * 16 + ln;

    // ---- score tiles for this wave's (mt, nt) combo ----
    v8f su = {}; v8f sd = {};
#pragma unroll
    for (int ks = 0; ks < Dd / 32; ++ks) {
      const int d0 = ks * 32 + dchunk;
      v16h au = *(const v16h*)(yuB  + (size_t)mrow * Dd + d0);
      v16h bu = *(const v16h*)(xd_h + (size_t)ncol * Dd + d0);
      su = WMMA_F16(au, bu, su);
      v16h ad = *(const v16h*)(xuB  + (size_t)mrow * Dd + d0);
      v16h bd = *(const v16h*)(yd_h + (size_t)ncol * Dd + d0);
      sd = WMMA_F16(ad, bd, sd);
    }

    // column statistics for this lane's column (same for both lane halves)
    const size_t co  = (size_t)b * Nn + ncol;
    const float cmu = cmax_u[co], isu = 1.f / csum_u[co];
    const float cmd = cmax_d[co], isd = 1.f / csum_d[co];

    // ---- p = exp(energy); stage to LDS + row-sum accumulation ----
#pragma unroll
    for (int r = 0; r < 8; ++r) {
      const float e = __expf(su[r] - cmu) * isu + __expf(sd[r] - cmd) * isd;
      const float p = __expf(e);                 // e in (0,2]: safe, no max pass
      const int mloc = mt * 16 + r + 8 * hi;     // C-layout row
      pLDS[mloc * 32 + nt * 16 + ln] = (_Float16)p;
      float pv = p;                              // reduce over 16 n-lanes (half-wave)
#pragma unroll
      for (int off = 1; off < 16; off <<= 1) pv += __shfl_xor(pv, off, 32);
      if (ln == 0) atomicAdd(&rowsum[mloc], pv);
    }
    __syncthreads();

    // ---- acc[k,m] += w[k, n0:n0+32] * p^T ----
#pragma unroll
    for (int kti = 0; kti < 2; ++kti) {
      const int krow = wave * 32 + kti * 16 + ln;  // this wave owns k-tiles 2w,2w+1
      v16h aw = *(const v16h*)(wB + (size_t)krow * Nn + n0 + dchunk);
#pragma unroll
      for (int mt2 = 0; mt2 < 4; ++mt2) {
        v16h bp = *(const v16h*)(pLDS + (mt2 * 16 + ln) * 32 + dchunk);
        acc[kti][mt2] = WMMA_F16(aw, bp, acc[kti][mt2]);
      }
    }
    __syncthreads();   // protect pLDS before next chunk overwrites it
  }

  // ---- normalize by rowsum and write out[b][k][m] (m contiguous) ----
#pragma unroll
  for (int mt2 = 0; mt2 < 4; ++mt2) {
    const float rinv = 1.f / rowsum[mt2 * 16 + ln];
#pragma unroll
    for (int kti = 0; kti < 2; ++kti) {
#pragma unroll
      for (int r = 0; r < 8; ++r) {
        const int k = wave * 32 + kti * 16 + r + 8 * hi;
        out[((size_t)b * Kk + k) * Mm + mbase + mt2 * 16 + ln] =
            acc[kti][mt2][r] * rinv;
      }
    }
  }
}

// ---------------------------------------------------------------------------
extern "C" void kernel_launch(void* const* d_in, const int* in_sizes, int n_in,
                              void* d_out, int out_size, void* d_ws, size_t ws_size,
                              hipStream_t stream)
{
  const float* yu = (const float*)d_in[0];
  const float* xu = (const float*)d_in[1];
  const float* xd = (const float*)d_in[2];
  const float* yd = (const float*)d_in[3];
  const float* vu = (const float*)d_in[4];
  const float* vd = (const float*)d_in[5];
  float* out = (float*)d_out;

  char* ws = (char*)d_ws;
  const size_t MiB = 1024ull * 1024ull;
  _Float16* yu_h = (_Float16*)(ws);                 //  8 MiB
  _Float16* xu_h = (_Float16*)(ws +  8 * MiB);      //  8 MiB
  _Float16* xd_h = (_Float16*)(ws + 16 * MiB);      //  1 MiB
  _Float16* yd_h = (_Float16*)(ws + 17 * MiB);      //  1 MiB
  _Float16* w_h  = (_Float16*)(ws + 18 * MiB);      //  8 MiB
  float* cmax_u  = (float*)(ws + 26 * MiB);         //  4 x 64 KiB stats
  float* csum_u  = cmax_u + (size_t)Bb * Nn;
  float* cmax_d  = csum_u + (size_t)Bb * Nn;
  float* csum_d  = cmax_d + (size_t)Bb * Nn;

  const int prep_elems = Bb * Mm * Dd;              // 4,194,304
  prep_kernel<<<(prep_elems + 255) / 256, 256, 0, stream>>>(
      yu, xu, xd, yd, vu, vd, yu_h, xu_h, xd_h, yd_h, w_h);

  colstats_kernel<<<dim3(Nn / 128, Bb), 256, 0, stream>>>(
      yu_h, xu_h, xd_h, yd_h, cmax_u, csum_u, cmax_d, csum_d);

  fused_kernel<<<dim3(Mm / 64, Bb), 256, 0, stream>>>(
      yu_h, xu_h, xd_h, yd_h, w_h, cmax_u, csum_u, cmax_d, csum_d, out);
}
